// RelationalMSG_88141318848530
// MI455X (gfx1250) — matrix operand order repaired and checked by reading, not compiled, not run on hardware
//
#include <hip/hip_runtime.h>

typedef __attribute__((ext_vector_type(16))) __bf16 v16bf;
typedef __attribute__((ext_vector_type(8)))  __bf16 v8bf;
typedef __attribute__((ext_vector_type(4)))  __bf16 v4bf;
typedef __attribute__((ext_vector_type(8)))  float  v8f;
typedef __attribute__((ext_vector_type(4)))  float  v4f;

constexpr int kNodes  = 50000;
constexpr int kEdges  = 600000;
constexpr int kD      = 128;          // feature dim
constexpr int kTilesE = kEdges / 16;  // 37500
constexpr int kTilesN = kNodes / 16;  // 3125
constexpr int kWPB    = 4;            // waves per block (blockDim = 128)
constexpr int kRow    = 272;          // padded bf16 row stride (256 + 32B pad)
constexpr int kWElems = 8 * 8 * 32 * 16;  // 32768 bf16 per swizzled weight matrix

// ---------------------------------------------------------------- zero agg/out
__global__ void zero_f32(float* __restrict__ p, int n) {
  int i = blockIdx.x * blockDim.x + threadIdx.x;
  int stride = gridDim.x * blockDim.x;
  for (; i < n; i += stride) p[i] = 0.0f;
}

// ------------------------------------------------- swizzle W1,W2 -> bf16 frags
// B-fragment layout per (kc,n) tile: 32 lanes x 16 contiguous bf16 per lane.
// element e of lane l maps to W[K][N], K = kc*32 + 8*(l>>4) + e + (e&8), N = n*16 + (l&15)
__global__ void swizzle_weights(const float* __restrict__ W1,
                                const float* __restrict__ W2,
                                __bf16* __restrict__ ws) {
  int i = blockIdx.x * blockDim.x + threadIdx.x;
  int stride = gridDim.x * blockDim.x;
  for (; i < 2 * kWElems; i += stride) {
    const float* W = (i < kWElems) ? W1 : W2;
    int t    = i & (kWElems - 1);
    int e    = t & 15;
    int lane = (t >> 4) & 31;
    int n    = (t >> 9) & 7;
    int kc   = (t >> 12) & 7;
    int K = kc * 32 + ((lane >> 4) << 3) + e + (e & 8);
    int N = n * 16 + (lane & 15);
    ws[i] = (__bf16)W[K * kD + N];
  }
}

__device__ __forceinline__ v16bf load_a_frag(const __bf16* rowPtr, int k0) {
  // A 16x32 bf16 fragment per lane: two contiguous 8-element runs at k0, k0+16
  v8bf lo = *(const v8bf*)(rowPtr + k0);
  v8bf hi = *(const v8bf*)(rowPtr + k0 + 16);
  v16bf a;
#pragma unroll
  for (int j = 0; j < 8; ++j) { a[j] = lo[j]; a[j + 8] = hi[j]; }
  return a;
}

// ------------------------------------------------------------------ edge MLP
__global__ __launch_bounds__(128) void edge_msg_kernel(
    const float* __restrict__ x, const int* __restrict__ esrc,
    const int* __restrict__ edst, const float* __restrict__ b1,
    const __bf16* __restrict__ wsW1, float* __restrict__ agg) {
  __shared__ __bf16 pairTile[kWPB][16][kRow];

  const int lane = threadIdx.x & 31;
  const int w    = threadIdx.x >> 5;
  const int hf   = lane >> 4;
  const int l15  = lane & 15;
  const int waveId = blockIdx.x * kWPB + w;
  const int nWaves = gridDim.x * kWPB;

  float bias[8];
#pragma unroll
  for (int n = 0; n < 8; ++n) bias[n] = b1[n * 16 + l15];

  for (int tile = waveId; tile < kTilesE; tile += nWaves) {
    const int eBase = tile * 16;
    // lanes 0-15 hold src indices of rows 0-15; lanes 16-31 hold dst indices
    int myIdx = (lane < 16) ? esrc[eBase + l15] : edst[eBase + l15];

    // gather 16 rows of [x[src] ; x[dst]] into LDS (bf16), coalesced per row
#pragma unroll 4
    for (int r = 0; r < 16; ++r) {
      int sIdx = __shfl(myIdx, r, 32);
      int dIdx = __shfl(myIdx, 16 + r, 32);
      v4f xs = *(const v4f*)(x + (size_t)sIdx * kD + lane * 4);
      v4f xd = *(const v4f*)(x + (size_t)dIdx * kD + lane * 4);
      v4bf bs, bd;
#pragma unroll
      for (int j = 0; j < 4; ++j) { bs[j] = (__bf16)xs[j]; bd[j] = (__bf16)xd[j]; }
      *(v4bf*)&pairTile[w][r][lane * 4]       = bs;
      *(v4bf*)&pairTile[w][r][kD + lane * 4]  = bd;
    }

    v8f acc[8] = {};
    const __bf16* rowPtr = &pairTile[w][l15][0];
#pragma unroll
    for (int kc = 0; kc < 8; ++kc) {
      v16bf a = load_a_frag(rowPtr, kc * 32 + hf * 8);
#pragma unroll
      for (int n = 0; n < 8; ++n) {
        v16bf b = *(const v16bf*)(wsW1 + (((kc * 8 + n) * 32 + lane) << 4));
        acc[n] = __builtin_amdgcn_wmma_f32_16x16x32_bf16(
            false, a, false, b, (short)0, acc[n], false, false);
      }
    }

    // bias + relu, scatter-add to agg[dst]
    int dstv[8];
#pragma unroll
    for (int v = 0; v < 8; ++v) dstv[v] = __shfl(myIdx, 16 + v + 8 * hf, 32);
#pragma unroll
    for (int n = 0; n < 8; ++n) {
      const int col = n * 16 + l15;
#pragma unroll
      for (int v = 0; v < 8; ++v) {
        float m = acc[n][v] + bias[n];
        m = m > 0.0f ? m : 0.0f;
        atomicAdd(&agg[(size_t)dstv[v] * kD + col], m);
      }
    }
  }
}

// ------------------------------------------------------------------ node MLP
__global__ __launch_bounds__(128) void node_update_kernel(
    const float* __restrict__ x, const float* __restrict__ b2,
    const __bf16* __restrict__ wsW2, float* __restrict__ out /* holds agg */) {
  __shared__ __bf16 pairTile[kWPB][16][kRow];

  const int lane = threadIdx.x & 31;
  const int w    = threadIdx.x >> 5;
  const int hf   = lane >> 4;
  const int l15  = lane & 15;
  const int waveId = blockIdx.x * kWPB + w;
  const int nWaves = gridDim.x * kWPB;

  float bias[8];
#pragma unroll
  for (int n = 0; n < 8; ++n) bias[n] = b2[n * 16 + l15];

  for (int tile = waveId; tile < kTilesN; tile += nWaves) {
    const int nBase = tile * 16;
#pragma unroll 4
    for (int r = 0; r < 16; ++r) {
      size_t row = (size_t)(nBase + r) * kD;
      v4f xs = *(const v4f*)(x + row + lane * 4);
      v4f ag = *(const v4f*)(out + row + lane * 4);   // agg from edge kernel
      v4bf bs, bd;
#pragma unroll
      for (int j = 0; j < 4; ++j) { bs[j] = (__bf16)xs[j]; bd[j] = (__bf16)ag[j]; }
      *(v4bf*)&pairTile[w][r][lane * 4]      = bs;
      *(v4bf*)&pairTile[w][r][kD + lane * 4] = bd;
    }

    v8f acc[8] = {};
    const __bf16* rowPtr = &pairTile[w][l15][0];
#pragma unroll
    for (int kc = 0; kc < 8; ++kc) {
      v16bf a = load_a_frag(rowPtr, kc * 32 + hf * 8);
#pragma unroll
      for (int n = 0; n < 8; ++n) {
        v16bf b = *(const v16bf*)(wsW2 + (((kc * 8 + n) * 32 + lane) << 4));
        acc[n] = __builtin_amdgcn_wmma_f32_16x16x32_bf16(
            false, a, false, b, (short)0, acc[n], false, false);
      }
    }

    // relu + residual, overwrite agg rows with final output
#pragma unroll
    for (int n = 0; n < 8; ++n) {
      const int col = n * 16 + l15;
#pragma unroll
      for (int v = 0; v < 8; ++v) {
        int node = nBase + v + 8 * hf;
        float u = acc[n][v] + bias[n];
        u = u > 0.0f ? u : 0.0f;
        out[(size_t)node * kD + col] = x[(size_t)node * kD + col] + u;
      }
    }
  }
}

extern "C" void kernel_launch(void* const* d_in, const int* in_sizes, int n_in,
                              void* d_out, int out_size, void* d_ws, size_t ws_size,
                              hipStream_t stream) {
  const float* x    = (const float*)d_in[0];
  const int*   esrc = (const int*)d_in[1];
  const int*   edst = (const int*)d_in[2];
  const float* W1   = (const float*)d_in[3];
  const float* b1   = (const float*)d_in[4];
  const float* W2   = (const float*)d_in[5];
  const float* b2   = (const float*)d_in[6];
  float*  out = (float*)d_out;
  __bf16* ws  = (__bf16*)d_ws;  // 2 * 32768 bf16 = 128 KB of swizzled weights

  (void)in_sizes; (void)n_in; (void)out_size; (void)ws_size;

  zero_f32<<<256, 256, 0, stream>>>(out, kNodes * kD);          // out doubles as agg
  swizzle_weights<<<256, 256, 0, stream>>>(W1, W2, ws);
  edge_msg_kernel<<<1024, 128, 0, stream>>>(x, esrc, edst, b1, ws, out);
  node_update_kernel<<<512, 128, 0, stream>>>(x, b2, ws + kWElems, out);
}
